// DiT_89086211653924
// MI455X (gfx1250) — compile-verified, matrix operand order checked by reading, not run on hardware
//
#include <hip/hip_runtime.h>
#include <hip/hip_bf16.h>
#include <math.h>

// ---------------- problem constants ----------------
constexpr int kB   = 2;
constexpr int kCIN = 3;
constexpr int kF   = 16;
constexpr int kIMG = 64;
constexpr int kP   = 4;
constexpr int kPD  = 16;   // IMG/P
constexpr int kNP  = 256;  // PD*PD
constexpr int kN   = 4096; // F*NP
constexpr int kC   = 512;
constexpr int kNH  = 8;
constexpr int kHD  = 64;
constexpr int kS   = 16;
constexpr int kM   = kB * kN;         // 8192 token rows
constexpr int kOUTC = kP * kP * kCIN; // 48
constexpr float kSCALE = 0.125f;      // 1/sqrt(64)

typedef __attribute__((ext_vector_type(16))) __bf16   v16bf;
typedef __attribute__((ext_vector_type(8)))  float    v8f;
typedef __attribute__((ext_vector_type(8)))  unsigned u32x8;

#define DEV static __device__ __forceinline__

DEV unsigned short f2bf(float f) {
  unsigned u = __builtin_bit_cast(unsigned, f);
  unsigned r = (u + 0x7FFFu + ((u >> 16) & 1u)) >> 16;
  return (unsigned short)r;
}
DEV unsigned pack2(float lo, float hi) {
  return (unsigned)f2bf(lo) | ((unsigned)f2bf(hi) << 16);
}
DEV v8f zero8() { v8f z = {0.f,0.f,0.f,0.f,0.f,0.f,0.f,0.f}; return z; }
DEV v8f wmma_bf16(u32x8 a, u32x8 b, v8f c) {
  return __builtin_amdgcn_wmma_f32_16x16x32_bf16(
      false, __builtin_bit_cast(v16bf, a),
      false, __builtin_bit_cast(v16bf, b),
      (short)0, c, false, false);
}
// A/B fragment K index pattern for 16x16x32 bf16 (per CDNA5 ISA 7.12.2)
DEV int frag_k(int ks, int i, int lane) {
  return ks * 32 + ((i >> 2) << 4) + (((lane >> 4) & 1) << 3) + ((i & 3) << 1);
}
DEV void lds_fence() {
  asm volatile("s_wait_dscnt 0x0" ::: "memory");
  __builtin_amdgcn_wave_barrier();
}

// ---------------- weight pre-pack (f32 -> bf16 fragment order) ----------------
// out[((ks*N16 + nt)*32 + lane)*8 + i] = pack(W[k,n], W[k+1,n]); n = nt*16 + lane&15
// (8 contiguous dwords per lane -> 2x b128 loads in the GEMM)
__global__ void pack_w_kernel(const float* __restrict__ W, unsigned* __restrict__ out,
                              int K, int N, int N16) {
  int t = blockIdx.x * 256 + threadIdx.x;
  int total = (K >> 5) * N16 * 256;
  if (t >= total) return;
  int i    = t & 7;
  int lane = (t >> 3) & 31;
  int rest = t >> 8;
  int nt   = rest % N16;
  int ks   = rest / N16;
  int k = frag_k(ks, i, lane);
  int n = nt * 16 + (lane & 15);
  out[t] = pack2(W[k * N + n], W[(k + 1) * N + n]);
}

// ---------------- patch embed + positional ----------------
__global__ void embed_kernel(const float* __restrict__ x, const float* __restrict__ cw,
                             const float* __restrict__ cb, const float* __restrict__ tpos,
                             const float* __restrict__ sppos, float* __restrict__ h) {
  int idx = blockIdx.x * 256 + threadIdx.x;
  if (idx >= kM * kC) return;
  int c = idx & (kC - 1);
  int n = (idx >> 9) & (kN - 1);
  int b = idx >> 21;
  int f  = n >> 8;
  int py = (n >> 4) & 15;
  int px = n & 15;
  float acc = cb[c];
#pragma unroll
  for (int ci = 0; ci < kCIN; ci++)
#pragma unroll
    for (int p = 0; p < kP; p++)
#pragma unroll
      for (int q = 0; q < kP; q++)
        acc += x[(((b * kCIN + ci) * kF + f) * kIMG + (py * kP + p)) * kIMG + (px * kP + q)] *
               cw[((c * kCIN + ci) * kP + p) * kP + q];
  acc += tpos[f * kC + c] + sppos[(py * kPD + px) * kC + c];
  h[idx] = acc;
}

// ---------------- action context embed (bf16) ----------------
__global__ void ctx_kernel(const int* __restrict__ actions, const float* __restrict__ table,
                           const float* __restrict__ apos, unsigned short* __restrict__ ctxb) {
  int idx = blockIdx.x * 256 + threadIdx.x;
  if (idx >= kB * kS * kC) return;
  int c = idx & (kC - 1);
  int s = (idx >> 9) & (kS - 1);
  int b = idx >> 13;
  int a = actions[b * kS + s];
  ctxb[idx] = f2bf(table[a * kC + c] + apos[s * kC + c]);
}

// ---------------- LayerNorm -> bf16 ----------------
__global__ void ln_kernel(const float* __restrict__ x, const float* __restrict__ g,
                          const float* __restrict__ bt, unsigned short* __restrict__ out) {
  __shared__ float s1[256];
  __shared__ float s2[256];
  int row = blockIdx.x;
  int t = threadIdx.x;
  const float* xr = x + (long)row * kC;
  float a0 = xr[t], a1 = xr[t + 256];
  s1[t] = a0 + a1;
  s2[t] = a0 * a0 + a1 * a1;
  __syncthreads();
  for (int off = 128; off; off >>= 1) {
    if (t < off) { s1[t] += s1[t + off]; s2[t] += s2[t + off]; }
    __syncthreads();
  }
  float mean = s1[0] * (1.f / kC);
  float var  = s2[0] * (1.f / kC) - mean * mean;
  float inv  = rsqrtf(var + 1e-5f);
  out[(long)row * kC + t]       = f2bf((a0 - mean) * inv * g[t] + bt[t]);
  out[(long)row * kC + t + 256] = f2bf((a1 - mean) * inv * g[t + 256] + bt[t + 256]);
}

// ---------------- generic WMMA GEMM: out = act(A @ Wp + bias) (+ res) ----------------
// A: bf16 row-major MxK (K=512). Wp: fragment-packed bf16 weights.
// One wave -> 32 x (NT*16) tile; software-pipelined (prefetch next K-step).
// outB: bf16 row-major [M x N]. outBT: bf16 transposed [N x M] (for attention V).
template<int NT>
__global__ __launch_bounds__(256) void gemm_kernel(
    const unsigned short* __restrict__ A, const unsigned* __restrict__ Wp,
    const float* __restrict__ bias, const float* __restrict__ res,
    float* __restrict__ outF, unsigned short* __restrict__ outB,
    unsigned short* __restrict__ outBT,
    int M, int N, int N16, int nTN, int doGelu) {
  constexpr int K = kC;
  constexpr int nKS = K >> 5;  // 16
  int lane = threadIdx.x & 31;
  int wid  = __builtin_amdgcn_readfirstlane(blockIdx.x * 8 + (threadIdx.x >> 5));
  int nTM  = M >> 5;          // 32-row tiles
  int tm   = wid / nTN;
  int tn   = wid % nTN;
  if (tm >= nTM) return;
  int m0 = tm << 5;
  int ml = lane & 15;
  const unsigned* Arow0 = (const unsigned*)(A + (size_t)(m0 + ml) * K);
  const unsigned* Arow1 = (const unsigned*)(A + (size_t)(m0 + 16 + ml) * K);
  v8f acc[2][NT];
#pragma unroll
  for (int mt = 0; mt < 2; mt++)
#pragma unroll
    for (int j = 0; j < NT; j++) acc[mt][j] = zero8();

  auto loadA = [&](int ks, u32x8& a0, u32x8& a1) {
#pragma unroll
    for (int i = 0; i < 8; i++) {
      int o = frag_k(ks, i, lane) >> 1;
      a0[i] = Arow0[o];
      a1[i] = Arow1[o];
    }
  };
  auto loadB = [&](int ks, u32x8* bu) {
#pragma unroll
    for (int j = 0; j < NT; j++) {
      long base8 = (((long)(ks * N16 + tn * NT + j)) * 32 + lane) * 8;
#pragma unroll
      for (int i = 0; i < 8; i++) bu[j][i] = Wp[base8 + i];
    }
  };
  auto compute = [&](u32x8& a0, u32x8& a1, u32x8* bu) {
#pragma unroll
    for (int j = 0; j < NT; j++) {
      acc[0][j] = wmma_bf16(a0, bu[j], acc[0][j]);
      acc[1][j] = wmma_bf16(a1, bu[j], acc[1][j]);
    }
  };

  // two-stage ping-pong software pipeline over K-steps
  u32x8 a0A, a1A, buA[NT], a0B, a1B, buB[NT];
  loadA(0, a0A, a1A);
  loadB(0, buA);
#pragma unroll 1
  for (int ks = 0; ks < nKS; ks += 2) {
    loadA(ks + 1, a0B, a1B);
    loadB(ks + 1, buB);
    compute(a0A, a1A, buA);
    if (ks + 2 < nKS) {
      loadA(ks + 2, a0A, a1A);
      loadB(ks + 2, buA);
    }
    compute(a0B, a1B, buB);
  }

#pragma unroll
  for (int mt = 0; mt < 2; mt++) {
#pragma unroll
    for (int j = 0; j < NT; j++) {
      int n = (tn * NT + j) * 16 + ml;
      float bsv = bias ? bias[n] : 0.f;
#pragma unroll
      for (int r = 0; r < 8; r++) {
        int m = m0 + mt * 16 + r + (((lane >> 4) & 1) << 3);
        float v = acc[mt][j][r] + bsv;
        if (doGelu) v = 0.5f * v * (1.f + erff(v * 0.70710678f));
        if (res)   v += res[(long)m * N + n];
        if (outF)  outF[(long)m * N + n] = v;
        if (outB)  outB[(long)m * N + n] = f2bf(v);
        if (outBT) outBT[(long)n * M + m] = f2bf(v);
      }
    }
  }
}

// ---------------- windowed attention ----------------
// part 0: spatial (window = frame, L=256). part 1: spatiotemporal 4x4x4 windows (L=64).
DEV int tok_index(int part, int w, int l) {
  if (part == 0) return w * kNP + l;
  int tb = w >> 4, hb = (w >> 2) & 3, wb = w & 3;
  int tw = l >> 4, hw = (l >> 2) & 3, ww = l & 3;
  int f = tb * 4 + tw, y = hb * 4 + hw, xc = wb * 4 + ww;
  return (f * kPD + y) * kPD + xc;
}

// one wave per (b, window, head, 16-row q tile).
// LDS per wave: 16 x Sctx f32 scores + 16 x Sctx/2 packed bf16 probabilities.
// vb is TRANSPOSED: vb[channel * vLd + token_row].
__global__ __launch_bounds__(128) void attn_kernel(
    const unsigned short* __restrict__ qb, const unsigned short* __restrict__ kb,
    const unsigned short* __restrict__ vb, unsigned short* __restrict__ ob,
    int part, int nWin, int Lwin, int Sctx, int cross, int vLd) {
  extern __shared__ char smem[];
  __shared__ float red[128];
  int lane  = threadIdx.x & 31;
  int wslot = threadIdx.x >> 5;
  float*    sc = (float*)smem + wslot * 16 * Sctx;
  unsigned* pb = (unsigned*)(smem + 4 * 16 * Sctx * 4) + wslot * 16 * (Sctx >> 1);
  int nQT = Lwin >> 4;
  int wid = __builtin_amdgcn_readfirstlane(blockIdx.x * 4 + wslot);
  int qt = wid % nQT; wid /= nQT;
  int h  = wid % kNH; wid /= kNH;
  int w  = wid % nWin;
  int b  = wid / nWin;
  int ml = lane & 15;

  // Q fragments for this 16-row tile (K dim = HD = 64 -> 2 steps)
  int qrow = b * kN + tok_index(part, w, qt * 16 + ml);
  const unsigned* qp = (const unsigned*)(qb + (long)qrow * kC + h * kHD);
  u32x8 qa[2];
#pragma unroll
  for (int ks = 0; ks < 2; ks++)
#pragma unroll
    for (int i = 0; i < 8; i++) qa[ks][i] = qp[frag_k(ks, i, lane) >> 1];

  // pass 1: scores = scale * Q K^T  (ping-pong prefetch of K tiles)
  int nST = Sctx >> 4;
  auto loadK = [&](int st, u32x8* bu) {
    int s = st * 16 + ml;
    int krow = cross ? (b * kS + s) : (b * kN + tok_index(part, w, s));
    const unsigned* kp = (const unsigned*)(kb + (long)krow * kC + h * kHD);
#pragma unroll
    for (int ks = 0; ks < 2; ks++)
#pragma unroll
      for (int i = 0; i < 8; i++) bu[ks][i] = kp[frag_k(ks, i, lane) >> 1];
  };
  auto scoreTile = [&](int st, u32x8* bu) {
    v8f acc = zero8();
#pragma unroll
    for (int ks = 0; ks < 2; ks++) acc = wmma_bf16(qa[ks], bu[ks], acc);
#pragma unroll
    for (int r = 0; r < 8; r++)
      sc[(r + (((lane >> 4) & 1) << 3)) * Sctx + st * 16 + ml] = acc[r] * kSCALE;
  };
  {
    u32x8 kA[2], kB2[2];
    loadK(0, kA);
#pragma unroll 1
    for (int st = 0; st < nST; st += 2) {
      if (st + 1 < nST) loadK(st + 1, kB2);
      scoreTile(st, kA);
      if (st + 2 < nST) loadK(st + 2, kA);
      if (st + 1 < nST) scoreTile(st + 1, kB2);
    }
  }
  lds_fence();

  // softmax: all 32 lanes; lane handles half of row (lane&15)
  {
    int half = lane >> 4;
    int s0 = half * (Sctx >> 1), s1 = s0 + (Sctx >> 1);
    float* rp = sc + ml * Sctx;
    float mx = -1e30f;
    for (int s = s0; s < s1; s++) mx = fmaxf(mx, rp[s]);
    red[wslot * 32 + lane] = mx;
    lds_fence();
    mx = fmaxf(red[wslot * 32 + ml], red[wslot * 32 + 16 + ml]);
    float sum = 0.f;
    for (int s = s0; s < s1; s++) { float e = __expf(rp[s] - mx); rp[s] = e; sum += e; }
    lds_fence();
    red[wslot * 32 + lane] = sum;
    lds_fence();
    sum = red[wslot * 32 + ml] + red[wslot * 32 + 16 + ml];
    float inv = 1.f / sum;
    // pack normalized probabilities as bf16 pairs
    for (int s = s0; s < s1; s += 2)
      pb[ml * (Sctx >> 1) + (s >> 1)] = pack2(rp[s] * inv, rp[s + 1] * inv);
  }
  lds_fence();

  // pass 2: O = P V  (K dim = Sctx, zero-padded to multiple of 32)
  int nKS = (Sctx + 31) >> 5;
#pragma unroll
  for (int dt = 0; dt < 4; dt++) {
    const unsigned* vcol = (const unsigned*)(vb + (size_t)(h * kHD + dt * 16 + ml) * vLd);
    v8f acc = zero8();
    for (int ks = 0; ks < nKS; ks++) {
      u32x8 au, bu;
#pragma unroll
      for (int i = 0; i < 8; i++) {
        int k = frag_k(ks, i, lane);
        au[i] = (k < Sctx) ? pb[ml * (Sctx >> 1) + (k >> 1)] : 0u;
        unsigned bv = 0;
        if (k < Sctx) {
          int row = cross ? (b * kS + k) : (b * kN + tok_index(part, w, k));
          bv = vcol[row >> 1];  // rows k and k+1 are adjacent tokens
        }
        bu[i] = bv;
      }
      acc = wmma_bf16(au, bu, acc);
    }
#pragma unroll
    for (int r = 0; r < 8; r++) {
      int l = qt * 16 + r + (((lane >> 4) & 1) << 3);
      int orow = b * kN + tok_index(part, w, l);
      ob[(long)orow * kC + h * kHD + dt * 16 + ml] = f2bf(acc[r]);
    }
  }
}

// ---------------- misc ----------------
__global__ void cvt_kernel(const float* __restrict__ in, unsigned short* __restrict__ out, int n) {
  int idx = blockIdx.x * 256 + threadIdx.x;
  if (idx < n) out[idx] = f2bf(in[idx]);
}

__global__ void unpatch_kernel(const float* __restrict__ t, float* __restrict__ out) {
  int idx = blockIdx.x * 256 + threadIdx.x;
  if (idx >= kB * kCIN * kF * kIMG * kIMG) return;
  int xc = idx & 63;
  int t1 = idx >> 6;
  int y  = t1 & 63;
  int t2 = t1 >> 6;
  int f  = t2 & 15;
  int t3 = t2 >> 4;
  int ci = t3 % kCIN;
  int b  = t3 / kCIN;
  int n = f * kNP + (y >> 2) * kPD + (xc >> 2);
  int inner = ((y & 3) * kP + (xc & 3)) * kCIN + ci;
  out[idx] = t[((long)(b * kN + n)) * kOUTC + inner];
}

// ---------------- host orchestration ----------------
extern "C" void kernel_launch(void* const* d_in, const int* in_sizes, int n_in,
                              void* d_out, int out_size, void* d_ws, size_t ws_size,
                              hipStream_t stream) {
  (void)in_sizes; (void)n_in; (void)out_size; (void)ws_size;
  const float* x        = (const float*)d_in[0];
  const int*   actions  = (const int*)d_in[1];
  const float* act_tab  = (const float*)d_in[2];
  const float* act_pos  = (const float*)d_in[3];
  const float* conv_w   = (const float*)d_in[4];
  const float* conv_b   = (const float*)d_in[5];
  const float* t_pos    = (const float*)d_in[6];
  const float* sp_pos   = (const float*)d_in[7];
  const float* head_w   = (const float*)d_in[8];
  const float* head_b   = (const float*)d_in[9];

  struct BW {
    const float *ln_g, *ln_b, *qkv1, *wo1, *bo1, *qkv2, *wo2, *bo2, *f1w, *f1b, *f2w, *f2b;
  } bw[2];
  for (int p = 0; p < 2; p++) {
    int a = 10 + p * 12;
    bw[p].ln_g = (const float*)d_in[a + 0];  bw[p].ln_b = (const float*)d_in[a + 1];
    bw[p].qkv1 = (const float*)d_in[a + 2];  bw[p].wo1  = (const float*)d_in[a + 3];
    bw[p].bo1  = (const float*)d_in[a + 4];  bw[p].qkv2 = (const float*)d_in[a + 5];
    bw[p].wo2  = (const float*)d_in[a + 6];  bw[p].bo2  = (const float*)d_in[a + 7];
    bw[p].f1w  = (const float*)d_in[a + 8];  bw[p].f1b  = (const float*)d_in[a + 9];
    bw[p].f2w  = (const float*)d_in[a + 10]; bw[p].f2b  = (const float*)d_in[a + 11];
  }

  // bump allocator over workspace
  char* wsp = (char*)d_ws;
  auto alloc = [&](size_t bytes) -> void* {
    void* r = wsp;
    wsp += (bytes + 255) & ~(size_t)255;
    return r;
  };
  float*          h    = (float*)alloc((size_t)kM * kC * 4);
  unsigned short* lnb  = (unsigned short*)alloc((size_t)kM * kC * 2);
  unsigned short* qb   = (unsigned short*)alloc((size_t)kM * kC * 2);
  unsigned short* kb   = (unsigned short*)alloc((size_t)kM * kC * 2);
  unsigned short* vT   = (unsigned short*)alloc((size_t)kM * kC * 2); // transposed V
  unsigned short* ab   = (unsigned short*)alloc((size_t)kM * kC * 2);
  unsigned short* ctxb = (unsigned short*)alloc((size_t)kB * kS * kC * 2);
  float*          t0   = (float*)alloc((size_t)kM * kOUTC * 4);

  auto pack = [&](const float* W, int K, int N) -> unsigned* {
    int N16 = N / 16;
    int total = (K / 32) * N16 * 256;
    unsigned* out = (unsigned*)alloc((size_t)total * 4);
    pack_w_kernel<<<(total + 255) / 256, 256, 0, stream>>>(W, out, K, N, N16);
    return out;
  };
  // pack all weights (bf16, fragment order)
  unsigned* wpk[2][10];
  for (int p = 0; p < 2; p++) {
    wpk[p][0] = pack(bw[p].qkv1 + 0 * kC * kC, kC, kC);  // q1
    wpk[p][1] = pack(bw[p].qkv1 + 1 * kC * kC, kC, kC);  // k1
    wpk[p][2] = pack(bw[p].qkv1 + 2 * kC * kC, kC, kC);  // v1
    wpk[p][3] = pack(bw[p].wo1, kC, kC);
    wpk[p][4] = pack(bw[p].qkv2 + 0 * kC * kC, kC, kC);  // q2
    wpk[p][5] = pack(bw[p].qkv2 + 1 * kC * kC, kC, kC);  // k2
    wpk[p][6] = pack(bw[p].qkv2 + 2 * kC * kC, kC, kC);  // v2
    wpk[p][7] = pack(bw[p].wo2, kC, kC);
    wpk[p][8] = pack(bw[p].f1w, kC, kC);
    wpk[p][9] = pack(bw[p].f2w, kC, kC);
  }
  unsigned* headp = pack(head_w, kC, kOUTC);

  // main GEMMs: N=512 -> N16=32, NT=4, nTN=8, 32-row M tiles
  auto gemm512 = [&](const unsigned short* A, const unsigned* Wp, const float* bias,
                     const float* res, float* outF, unsigned short* outB,
                     unsigned short* outBT, int M, int gelu) {
    int nTN = 8;
    int waves = (M >> 5) * nTN;
    gemm_kernel<4><<<(waves + 7) / 8, 256, 0, stream>>>(A, Wp, bias, res, outF, outB, outBT,
                                                        M, kC, 32, nTN, gelu);
  };
  // head GEMM: N=48 -> N16=3, NT=3, nTN=1
  auto gemm48 = [&](const unsigned short* A, const unsigned* Wp, const float* bias,
                    float* outF, int M) {
    int waves = (M >> 5);
    gemm_kernel<3><<<(waves + 7) / 8, 256, 0, stream>>>(A, Wp, bias, nullptr, outF, nullptr,
                                                        nullptr, M, kOUTC, 3, 1, 0);
  };
  auto attn = [&](const unsigned short* q, const unsigned short* k, const unsigned short* v,
                  unsigned short* o, int part, int nWin, int Lwin, int Sctx, int cross, int vLd) {
    int waves = kB * nWin * kNH * (Lwin >> 4);  // 4096 for every config here
    size_t shmem = (size_t)4 * 16 * Sctx * 4 + (size_t)4 * 16 * (Sctx / 2) * 4;
    attn_kernel<<<waves / 4, 128, shmem, stream>>>(q, k, v, o, part, nWin, Lwin, Sctx, cross, vLd);
  };

  // ---- patch embed + pos; action context ----
  embed_kernel<<<(kM * kC + 255) / 256, 256, 0, stream>>>(x, conv_w, conv_b, t_pos, sp_pos, h);
  ctx_kernel<<<(kB * kS * kC + 255) / 256, 256, 0, stream>>>(actions, act_tab, act_pos, ctxb);

  // ---- two transformer blocks ----
  for (int p = 0; p < 2; p++) {
    const BW& W = bw[p];
    int part = p;                      // 0: spatial, 1: spatiotemporal
    int nWin = (p == 0) ? kF : 64;
    int Lwin = (p == 0) ? kNP : 64;

    // self window attention
    ln_kernel<<<kM, 256, 0, stream>>>(h, W.ln_g + 0 * kC, W.ln_b + 0 * kC, lnb);
    gemm512(lnb, wpk[p][0], nullptr, nullptr, nullptr, qb, nullptr, kM, 0);
    gemm512(lnb, wpk[p][1], nullptr, nullptr, nullptr, kb, nullptr, kM, 0);
    gemm512(lnb, wpk[p][2], nullptr, nullptr, nullptr, nullptr, vT, kM, 0);
    attn(qb, kb, vT, ab, part, nWin, Lwin, /*Sctx=*/Lwin, /*cross=*/0, /*vLd=*/kM);
    gemm512(ab, wpk[p][3], W.bo1, /*res=*/h, /*outF=*/h, nullptr, nullptr, kM, 0);

    // cross attention to action context
    ln_kernel<<<kM, 256, 0, stream>>>(h, W.ln_g + 1 * kC, W.ln_b + 1 * kC, lnb);
    gemm512(lnb,  wpk[p][4], nullptr, nullptr, nullptr, qb, nullptr, kM,      0);
    gemm512(ctxb, wpk[p][5], nullptr, nullptr, nullptr, kb, nullptr, kB * kS, 0);
    gemm512(ctxb, wpk[p][6], nullptr, nullptr, nullptr, nullptr, vT, kB * kS, 0);
    attn(qb, kb, vT, ab, part, nWin, Lwin, /*Sctx=*/kS, /*cross=*/1, /*vLd=*/kB * kS);
    gemm512(ab, wpk[p][7], W.bo2, /*res=*/h, /*outF=*/h, nullptr, nullptr, kM, 0);

    // MLP (C -> C, exact GELU, C -> C) + residual
    ln_kernel<<<kM, 256, 0, stream>>>(h, W.ln_g + 2 * kC, W.ln_b + 2 * kC, lnb);
    gemm512(lnb, wpk[p][8], W.f1b, nullptr, nullptr, ab, nullptr, kM, /*gelu=*/1);
    gemm512(ab,  wpk[p][9], W.f2b, /*res=*/h, /*outF=*/h, nullptr, nullptr, kM, 0);
  }

  // ---- head + un-patchify ----
  cvt_kernel<<<(kM * kC + 255) / 256, 256, 0, stream>>>(h, lnb, kM * kC);
  gemm48(lnb, headp, head_b, t0, kM);
  unpatch_kernel<<<(kB * kCIN * kF * kIMG * kIMG + 255) / 256, 256, 0, stream>>>(t0, (float*)d_out);
}
